// MultiHeadAttention_19602230739034
// MI455X (gfx1250) — matrix operand order repaired and checked by reading, not compiled, not run on hardware
//
#include <hip/hip_runtime.h>
#include <cstdint>
#include <cstddef>

// ---------------------------------------------------------------------------
// MultiHeadAttention for MI455X (gfx1250, wave32, WMMA 16x16x32 bf16)
// B=8, T=1024, C=768, H=12, hd=64
//
//   k_cvt      : x fp32 -> bf16                      [8192][768]
//   k_cvt_tr   : W_attn -> bf16 transposed (N-major) [2304][768]
//   k_cvt_tr   : W_proj -> bf16 transposed (N-major) [768][768]
//   k_qkv      : GEMM, B tile staged by TDM (tensor_load_to_lds, double buf)
//   k_attn     : flash attention, WMMA S/PV, ds_swizzle softmax reductions
//   k_proj     : GEMM, B tile staged by global_load_async_to_lds_b128
// ---------------------------------------------------------------------------

typedef __attribute__((ext_vector_type(16))) __bf16    bf16x16;
typedef __attribute__((ext_vector_type(8)))  float     f32x8;
typedef __attribute__((ext_vector_type(4)))  unsigned  u32x4;
typedef __attribute__((ext_vector_type(8)))  int       i32x8;
typedef __attribute__((ext_vector_type(4)))  int       i32x4;

union Frag {
  bf16x16 v;
  u32x4   q[2];
};

__device__ __forceinline__ unsigned short f2bf(float f) {
  unsigned u = __float_as_uint(f);
  u += 0x7FFFu + ((u >> 16) & 1u);        // round-to-nearest-even
  return (unsigned short)(u >> 16);
}

__device__ __forceinline__ f32x8 wmma_bf16(const Frag& a, const Frag& b, f32x8 c) {
  return __builtin_amdgcn_wmma_f32_16x16x32_bf16(false, a.v, false, b.v,
                                                 (short)0, c, false, false);
}

// The TDM / async-LDS instructions are not modeled as LDS writers. Capture the
// LDS pointer into an asm operand (so alias analysis must assume asm can write
// it), then the per-iteration memory-clobber asm acts as a potential writer.
#define LDS_CAPTURE(p)                                            \
  do { const void* _cap = (const void*)(p);                       \
       asm volatile("" :: "v"(_cap) : "memory"); } while (0)
#define LDS_WRITE_FENCE() asm volatile("" ::: "memory")

// group-of-32 ds_swizzle xor patterns: offset = (xor<<10) | and_mask(0x1f)
#define SWZ_MAXF(x, m) fmaxf((x), __int_as_float(__builtin_amdgcn_ds_swizzle(__float_as_int(x), (((m) << 10) | 0x1F))))
#define SWZ_ADDF(x, m) ((x) + __int_as_float(__builtin_amdgcn_ds_swizzle(__float_as_int(x), (((m) << 10) | 0x1F))))

// --- Tensor Data Mover: 2D tile load (64 rows x 32 cols bf16) into LDS -----
// D# per CDNA5 ISA: group0 {count, lds_addr,
// global_addr, type=2}, group1 {data_size=2B, tensor dims/strides, tile dims}.
__device__ __forceinline__ void tdm_load_tile(unsigned lds_off, const void* gbase,
                                              long elem_off, int tensor_dim1) {
  unsigned long long ga = (unsigned long long)gbase + (unsigned long long)elem_off * 2ull;
  u32x4 g0;
  g0.x = 1u;                                   // [1:0] count=1 (valid D#)
  g0.y = lds_off;                              // [63:32] lds_addr (bytes)
  g0.z = (unsigned)ga;                         // [95:64] global_addr lo
  g0.w = ((unsigned)(ga >> 32) & 0x01FFFFFFu)  // [120:96] global_addr hi
         | 0x80000000u;                        // [127:126] type=2 ("image")
  i32x8 g1;
  g1[0] = 0x00010000;                          // data_size=1 -> 2 bytes
  g1[1] = (768 & 0xFFFF) << 16;                // tensor_dim0 = 768 (K)
  g1[2] = (tensor_dim1 & 0xFFFF) << 16;        // tensor_dim1 (N rows)
  g1[3] = 32 << 16;                            // tile_dim0 = 32 (K slab)
  g1[4] = 64;                                  // tile_dim1 = 64 rows, tile_dim2=0
  g1[5] = 768;                                 // tensor_dim0_stride lo
  g1[6] = 0;                                   // stride hi / dim1_stride lo
  g1[7] = 0;                                   // dim1_stride hi
  i32x4 z4 = {0, 0, 0, 0};
  i32x8 z8 = {0, 0, 0, 0, 0, 0, 0, 0};
  __builtin_amdgcn_tensor_load_to_lds(g0, g1, z4, z4, z8, 0);
}

// -------------------------- conversion kernels -----------------------------

__global__ void k_cvt(const float* __restrict__ in, unsigned short* __restrict__ out, int n) {
  int i = blockIdx.x * 256 + threadIdx.x;
  if (i < n) out[i] = f2bf(in[i]);
}

// out[r][c] = in[c][r];  in is [cols_out][rows_out]
__global__ void k_cvt_tr(const float* __restrict__ in, unsigned short* __restrict__ out,
                         int rows_out, int cols_out) {
  int i = blockIdx.x * 256 + threadIdx.x;
  if (i < rows_out * cols_out) {
    int r = i / cols_out, c = i % cols_out;
    out[i] = f2bf(in[(long)c * rows_out + r]);
  }
}

// ------------------------------ QKV GEMM -----------------------------------
// [8192 x 768] @ [768 x 2304]; block tile 64(M) x 64(N), 4 waves, K step 32.
// Weight tile DMA'd into LDS by the TDM, double-buffered (TENSORcnt).

__global__ void __launch_bounds__(128)
k_qkv(const u32x4* __restrict__ xb, const u32x4* __restrict__ wat,
      const float* __restrict__ bias,
      unsigned short* __restrict__ Qo, unsigned short* __restrict__ Ko,
      unsigned short* __restrict__ Vo) {
  __shared__ u32x4 sB[512];                 // 2 x (64 N-rows x 32 K bf16)
  const int tid = threadIdx.x;
  const int wid = tid >> 5, lane = tid & 31;
  const int ln = lane & 15, half = lane >> 4;
  const int m0 = blockIdx.x * 64 + wid * 16;
  const int n0 = blockIdx.y * 64;

  LDS_CAPTURE(sB);                          // sB may be written by DMA/asm

  f32x8 acc[4] = {};
  const int rowA = (m0 + ln) * 96;          // 768 bf16 per row = 96 u32x4

  if (wid == 0) tdm_load_tile(0u, wat, (long)n0 * 768, 2304);   // prime buf0

  for (int k0 = 0; k0 < 768; k0 += 32) {
    const int kq  = k0 >> 3;
    const int buf = (k0 >> 5) & 1;
    if (wid == 0) {
      if (k0 + 32 < 768) {                  // issue next slab into other buffer
        tdm_load_tile((unsigned)((buf ^ 1) * 4096), wat,
                      (long)n0 * 768 + (k0 + 32), 2304);
        __builtin_amdgcn_s_wait_tensorcnt(1);   // current slab landed
      } else {
        __builtin_amdgcn_s_wait_tensorcnt(0);
      }
    }
    __syncthreads();
    LDS_WRITE_FENCE();                      // potential writer of captured sB
    const u32x4* sBc = &sB[buf * 256];
    Frag a;                                 // A frag: lane = M, K range 16*half..+15
    a.q[0] = xb[rowA + kq + 2 * half];
    a.q[1] = xb[rowA + kq + 2 * half + 1];
    if (k0 + 32 < 768) __builtin_prefetch(&xb[rowA + kq + 4 + half], 0, 1);
#pragma unroll
    for (int nt = 0; nt < 4; ++nt) {        // B frag: lane = N, K range 16*half..+15
      Frag b;
      b.q[0] = sBc[(nt * 16 + ln) * 4 + 2 * half];
      b.q[1] = sBc[(nt * 16 + ln) * 4 + 2 * half + 1];
      acc[nt] = wmma_bf16(a, b, acc[nt]);
    }
    __syncthreads();
  }

  // Epilogue: D layout = lane col (ln), rows v + 8*half. Scatter to Q/K/Vt.
  const int which = n0 / 768;               // 0=Q 1=K 2=V
  const int h = (n0 % 768) / 64;
  unsigned short* __restrict__ outp = (which == 0) ? Qo : (which == 1) ? Ko : Vo;
#pragma unroll
  for (int nt = 0; nt < 4; ++nt) {
    const float bv = bias[n0 + nt * 16 + ln];
    const int d = nt * 16 + ln;
#pragma unroll
    for (int v = 0; v < 8; ++v) {
      const int mg = m0 + v + 8 * half;
      const int b = mg >> 10, t = mg & 1023;
      const int bh = b * 12 + h;
      const unsigned short val = f2bf(acc[nt][v] + bv);
      if (which == 2) outp[(bh * 64 + d) * 1024 + t] = val;   // Vt[BH][64][T]
      else            outp[(bh * 1024 + t) * 64 + d] = val;   // Q/K[BH][T][64]
    }
  }
}

// ---------------------------- flash attention ------------------------------
// One wave = one 16-query tile of one (b,h). Streams 32-key steps.

__global__ void __launch_bounds__(128)
k_attn(const u32x4* __restrict__ Qm, const u32x4* __restrict__ Km,
       const u32x4* __restrict__ Vm, unsigned short* __restrict__ AO) {
  __shared__ u32x4 sP[4][64];               // per-wave 16x32 bf16 P tile
  const int tid = threadIdx.x;
  const int wid = tid >> 5, lane = tid & 31;
  const int ln = lane & 15, half = lane >> 4;
  const int g  = blockIdx.x * 4 + wid;      // 6144 wave tasks total
  const int qt = g & 63;
  const int bh = g >> 6;
  const int q0 = qt * 16;

  // Q A-frags (d 0..31 and d 32..63), loaded once
  Frag aq0, aq1;
  {
    const int rq = (bh * 1024 + q0 + ln) * 8;   // 64 bf16 per row = 8 u32x4
    aq0.q[0] = Qm[rq + 2 * half];     aq0.q[1] = Qm[rq + 2 * half + 1];
    aq1.q[0] = Qm[rq + 2 * half + 4]; aq1.q[1] = Qm[rq + 2 * half + 5];
  }

  f32x8 o0 = {}, o1 = {}, o2 = {}, o3 = {};
  float rm[8], rs[8];
#pragma unroll
  for (int v = 0; v < 8; ++v) { rm[v] = -3.0e38f; rs[v] = 0.0f; }

  unsigned short* __restrict__ spw = (unsigned short*)&sP[wid][0];
  const u32x4*    __restrict__ spq = &sP[wid][0];

  const int nsteps = (q0 + 16 + 31) >> 5;   // causal: keys 0 .. q0+15
  for (int s = 0; s < nsteps; ++s) {
    const int k0 = s << 5;
    // ---- S = Q @ K^T for 32 keys (two 16x16 tiles) ----
    f32x8 Sa = {}, Sb = {};
    {
      const int ra = (bh * 1024 + k0 + ln) * 8;       // keys k0..k0+15
      Frag bk;
      bk.q[0] = Km[ra + 2 * half];     bk.q[1] = Km[ra + 2 * half + 1];
      Sa = wmma_bf16(aq0, bk, Sa);
      bk.q[0] = Km[ra + 2 * half + 4]; bk.q[1] = Km[ra + 2 * half + 5];
      Sa = wmma_bf16(aq1, bk, Sa);
      const int rb = ra + 128;                        // keys k0+16..k0+31
      bk.q[0] = Km[rb + 2 * half];     bk.q[1] = Km[rb + 2 * half + 1];
      Sb = wmma_bf16(aq0, bk, Sb);
      bk.q[0] = Km[rb + 2 * half + 4]; bk.q[1] = Km[rb + 2 * half + 5];
      Sb = wmma_bf16(aq1, bk, Sb);
    }
    // ---- online softmax (rows live across 16-lane groups) ----
    const float scale = 0.125f;                       // 1/sqrt(64)
    float fac[8];
#pragma unroll
    for (int v = 0; v < 8; ++v) {
      const int qrow = q0 + v + 8 * half;
      float sa = (k0 + ln      <= qrow) ? Sa[v] * scale : -3.0e38f;
      float sb = (k0 + 16 + ln <= qrow) ? Sb[v] * scale : -3.0e38f;
      float mx = fmaxf(sa, sb);
      mx = SWZ_MAXF(mx, 1); mx = SWZ_MAXF(mx, 2);
      mx = SWZ_MAXF(mx, 4); mx = SWZ_MAXF(mx, 8);
      const float nm = fmaxf(rm[v], mx);
      const float f  = __expf(rm[v] - nm);
      rm[v] = nm;
      const float pa = __expf(sa - nm);
      const float pb = __expf(sb - nm);
      float ps = pa + pb;
      ps = SWZ_ADDF(ps, 1); ps = SWZ_ADDF(ps, 2);
      ps = SWZ_ADDF(ps, 4); ps = SWZ_ADDF(ps, 8);
      rs[v] = rs[v] * f + ps;
      fac[v] = f;
      const int m = v + 8 * half;                     // P tile store (C/D layout)
      spw[m * 32 + ln]      = f2bf(pa);
      spw[m * 32 + 16 + ln] = f2bf(pb);
    }
#pragma unroll
    for (int v = 0; v < 8; ++v) {
      o0[v] *= fac[v]; o1[v] *= fac[v]; o2[v] *= fac[v]; o3[v] *= fac[v];
    }
    asm volatile("s_wait_dscnt 0" ::: "memory");      // P stores -> A-frag reload
    // ---- P A-frag (lane transpose via LDS) ----
    Frag ap;
    ap.q[0] = spq[ln * 4 + 2 * half];
    ap.q[1] = spq[ln * 4 + 2 * half + 1];
    // ---- O += P @ V  (Vt[BH][64][T]: column d is key-contiguous) ----
    const int kv = (k0 >> 3) + 2 * half;
    {
      Frag bv;
      int base = (bh * 64 + ln) * 128 + kv;
      bv.q[0] = Vm[base]; bv.q[1] = Vm[base + 1]; o0 = wmma_bf16(ap, bv, o0);
      base += 16 * 128;
      bv.q[0] = Vm[base]; bv.q[1] = Vm[base + 1]; o1 = wmma_bf16(ap, bv, o1);
      base += 16 * 128;
      bv.q[0] = Vm[base]; bv.q[1] = Vm[base + 1]; o2 = wmma_bf16(ap, bv, o2);
      base += 16 * 128;
      bv.q[0] = Vm[base]; bv.q[1] = Vm[base + 1]; o3 = wmma_bf16(ap, bv, o3);
    }
  }

  // ---- normalize + store bf16 to AO [B*T][768] ----
  const int b = bh / 12, h = bh % 12;
#pragma unroll
  for (int v = 0; v < 8; ++v) {
    const float inv = 1.0f / rs[v];
    const int row = b * 1024 + q0 + v + 8 * half;
    unsigned short* __restrict__ orow = AO + (long)row * 768 + h * 64;
    orow[ln]      = f2bf(o0[v] * inv);
    orow[16 + ln] = f2bf(o1[v] * inv);
    orow[32 + ln] = f2bf(o2[v] * inv);
    orow[48 + ln] = f2bf(o3[v] * inv);
  }
}

// ---------------------------- output projection ----------------------------
// B tile staged memory->LDS via global_load_async_to_lds_b128 (ASYNCcnt).

__global__ void __launch_bounds__(128)
k_proj(const u32x4* __restrict__ ao, const u32x4* __restrict__ wpt,
       const float* __restrict__ bias, float* __restrict__ out) {
  __shared__ u32x4 sB[256];                 // 64 N-rows x 32 K bf16, LDS off 0
  const int tid = threadIdx.x;
  const int wid = tid >> 5, lane = tid & 31;
  const int ln = lane & 15, half = lane >> 4;
  const int m0 = blockIdx.x * 64 + wid * 16;
  const int n0 = blockIdx.y * 64;

  LDS_CAPTURE(sB);                          // sB may be written by async/asm

  f32x8 acc[4] = {};
  const int rowA = (m0 + ln) * 96;
  for (int k0 = 0; k0 < 768; k0 += 32) {
    const int kq = k0 >> 3;
    {
      int idx = tid, r = idx >> 2, q = idx & 3;
      unsigned goff = (unsigned)(((n0 + r) * 96 + kq + q) * 16);
      asm volatile("global_load_async_to_lds_b128 %0, %1, %2"
                   :: "v"(idx * 16), "v"(goff), "s"(wpt) : "memory");
      idx += 128; r = idx >> 2; q = idx & 3;
      goff = (unsigned)(((n0 + r) * 96 + kq + q) * 16);
      asm volatile("global_load_async_to_lds_b128 %0, %1, %2"
                   :: "v"(idx * 16), "v"(goff), "s"(wpt) : "memory");
    }
    asm volatile("s_wait_asynccnt 0" ::: "memory");
    __syncthreads();
    LDS_WRITE_FENCE();
    Frag a;
    a.q[0] = ao[rowA + kq + 2 * half];
    a.q[1] = ao[rowA + kq + 2 * half + 1];
#pragma unroll
    for (int nt = 0; nt < 4; ++nt) {
      Frag b;
      b.q[0] = sB[(nt * 16 + ln) * 4 + 2 * half];
      b.q[1] = sB[(nt * 16 + ln) * 4 + 2 * half + 1];
      acc[nt] = wmma_bf16(a, b, acc[nt]);
    }
    __syncthreads();
  }
#pragma unroll
  for (int nt = 0; nt < 4; ++nt) {
    const int ng = n0 + nt * 16 + ln;
    const float bv = bias[ng];
#pragma unroll
    for (int v = 0; v < 8; ++v) {
      const int mg = m0 + v + 8 * half;
      out[(long)mg * 768 + ng] = acc[nt][v] + bv;
    }
  }
}

// ------------------------------- launcher ----------------------------------

extern "C" void kernel_launch(void* const* d_in, const int* in_sizes, int n_in,
                              void* d_out, int out_size, void* d_ws, size_t ws_size,
                              hipStream_t stream) {
  (void)in_sizes; (void)n_in; (void)out_size; (void)ws_size;
  const float* x      = (const float*)d_in[0];
  const float* W_attn = (const float*)d_in[1];
  const float* b_attn = (const float*)d_in[2];
  const float* W_proj = (const float*)d_in[3];
  const float* b_proj = (const float*)d_in[4];
  float* out = (float*)d_out;

  char* ws = (char*)d_ws;
  unsigned short* XB  = (unsigned short*)(ws);                 // 12,582,912 B
  unsigned short* WAT = (unsigned short*)(ws + 12582912);      //  3,538,944 B
  unsigned short* WPT = (unsigned short*)(ws + 16121856);      //  1,179,648 B
  unsigned short* Qb  = (unsigned short*)(ws + 17301504);      // 12,582,912 B
  unsigned short* Kb  = (unsigned short*)(ws + 29884416);      // 12,582,912 B
  unsigned short* Vb  = (unsigned short*)(ws + 42467328);      // 12,582,912 B
  unsigned short* AO  = (unsigned short*)(ws + 55050240);      // 12,582,912 B

  k_cvt   <<<(6291456 + 255) / 256, 256, 0, stream>>>(x, XB, 6291456);
  k_cvt_tr<<<(1769472 + 255) / 256, 256, 0, stream>>>(W_attn, WAT, 2304, 768);
  k_cvt_tr<<<(589824  + 255) / 256, 256, 0, stream>>>(W_proj, WPT, 768, 768);

  k_qkv <<<dim3(128, 36), 128, 0, stream>>>((const u32x4*)XB, (const u32x4*)WAT,
                                            b_attn, Qb, Kb, Vb);
  k_attn<<<1536, 128, 0, stream>>>((const u32x4*)Qb, (const u32x4*)Kb,
                                   (const u32x4*)Vb, AO);
  k_proj<<<dim3(128, 12), 128, 0, stream>>>((const u32x4*)AO, (const u32x4*)WPT,
                                            b_proj, out);
}
